// EGNN_41111426957350
// MI455X (gfx1250) — matrix operand order repaired and checked by reading, not compiled, not run on hardware
//
#include <hip/hip_runtime.h>
#include <hip/hip_bf16.h>

#define N_NODES 50000
#define N_EDGES 400000
#define NFEAT   32
#define HID     64

typedef __attribute__((ext_vector_type(16))) __bf16 v16bf;
typedef __attribute__((ext_vector_type(8)))  float  v8f;

// fast SiLU: v_exp_f32 + v_add + v_rcp_f32 + v_mul (avoids IEEE div expansion)
static __device__ __forceinline__ float siluf(float x) {
    return x * __builtin_amdgcn_rcpf(1.0f + __expf(-x));
}

static __device__ __forceinline__ unsigned pack2bf(float a, float b) {
    union { __bf16 h[2]; unsigned u; } x;
    x.h[0] = (__bf16)a; x.h[1] = (__bf16)b;
    return x.u;
}

// ---- A fragment from LDS row-major bf16 tile (16-bit 16x32 A layout, CDNA5 ISA) ----
static __device__ __forceinline__ v16bf load_a_lds(const __bf16* tile, int stride, int kc0, int lane) {
    const int half = (lane >> 4) & 1;
    const int m    = lane & 15;
    const __bf16* r = tile + m * stride + kc0 + 8 * half;
    v16bf a;
#pragma unroll
    for (int v = 0; v < 8; ++v) {
        const int kb = 2 * v + (v >= 4 ? 8 : 0);
        a[2 * v]     = r[kb];
        a[2 * v + 1] = r[kb + 1];
    }
    return a;
}

// ---- B fragment from pre-swizzled packed weights: one 32B load per lane ----
static __device__ __forceinline__ v16bf load_b_packed(const unsigned* __restrict__ Wp,
                                                      int fc, int wv, int lane) {
    const v16bf* p = (const v16bf*)(Wp + ((((fc * 4) + wv) * 32) + lane) * 8);
    return *p;
}

// Pre-swizzle W [K x 64] f32 -> packed bf16 B fragments, K zero-padded to KC*32.
__global__ void k_pack_weights(const float* __restrict__ W, int K, int KC, unsigned* __restrict__ out) {
    int t = blockIdx.x * blockDim.x + threadIdx.x;
    int total = KC * 1024;
    if (t >= total) return;
    int v = t & 7;
    int lane = (t >> 3) & 31;
    int fw = t >> 8;
    int w = fw & 3;
    int fc = fw >> 2;
    int half = lane >> 4;
    int n = (lane & 15) + w * 16;
    int k0 = fc * 32 + 8 * half + 2 * v + (v >= 4 ? 8 : 0);
    float f0 = (k0 < K) ? W[k0 * HID + n] : 0.0f;
    float f1 = (k0 + 1 < K) ? W[(k0 + 1) * HID + n] : 0.0f;
    out[t] = pack2bf(f0, f1);
}

// ---------------- small utility kernels ----------------
__global__ void k_fill(float* __restrict__ p, float v, int n) {
    int i = blockIdx.x * blockDim.x + threadIdx.x;
    if (i < n) p[i] = v;
}

__global__ void k_copy(float* __restrict__ dst, const float* __restrict__ src, int n) {
    int i = blockIdx.x * blockDim.x + threadIdx.x;
    if (i < n) dst[i] = src[i];
}

__global__ void k_count(const int* __restrict__ ei, float* __restrict__ deg, float* __restrict__ cnt) {
    int e = blockIdx.x * blockDim.x + threadIdx.x;
    if (e < N_EDGES) {
        atomicAdd(&deg[ei[N_EDGES + e]], 1.0f);  // dst degree (self loop preinit = 1)
        atomicAdd(&cnt[ei[e]], 1.0f);            // EGCL row count (row = src)
    }
}

__global__ void k_finalize(float* __restrict__ deg, float* __restrict__ cnt) {
    int i = blockIdx.x * blockDim.x + threadIdx.x;
    if (i < N_NODES) {
        deg[i] = rsqrtf(fmaxf(deg[i], 1.0f));  // becomes dinv
        cnt[i] = fmaxf(cnt[i], 1.0f);
    }
}

// ---------------- generic row-tile GEMM: out = act([A1|A2]@W + b) (+ residual) ----------------
// KC = K/32 chunks (compile-time). 16 rows/block, 4 waves cover 64 output cols.
template <int KC>
__global__ void __launch_bounds__(128) k_rowtile_gemm(
    const float* __restrict__ A1, int w1,
    const float* __restrict__ A2, int w2,
    const unsigned* __restrict__ Wp, const float* __restrict__ bias,
    const float* __restrict__ residual,
    float* __restrict__ out,
    int act_in_relu, int act_out_silu) {
    __shared__ __bf16 sA[16][132];
    const int r0 = blockIdx.x * 16;
    const int t  = threadIdx.x;
    {
        const int m = t >> 3, j = t & 7;
        const int row = r0 + m;
        unsigned* rp = (unsigned*)&sA[m][0];
        for (int k = j * 2; k < w1; k += 16) {
            float a = A1[row * w1 + k], b = A1[row * w1 + k + 1];
            if (act_in_relu) { a = fmaxf(a, 0.0f); b = fmaxf(b, 0.0f); }
            rp[k >> 1] = pack2bf(a, b);
        }
        for (int k = j * 2; k < w2; k += 16) {
            rp[(w1 + k) >> 1] = pack2bf(A2[row * w2 + k], A2[row * w2 + k + 1]);
        }
    }
    __syncthreads();

    const int lane = t & 31;
    const int wv   = t >> 5;
    const int n0   = wv * 16;
    const int half = lane >> 4, n = lane & 15;

    v8f acc;
    const float b = bias[n0 + n];
#pragma unroll
    for (int v = 0; v < 8; ++v) acc[v] = b;

#pragma unroll
    for (int fc = 0; fc < KC; ++fc) {
        v16bf a  = load_a_lds(&sA[0][0], 132, fc * 32, lane);
        v16bf bm = load_b_packed(Wp, fc, wv, lane);
        acc = __builtin_amdgcn_wmma_f32_16x16x32_bf16(false, a, false, bm, (short)0, acc, false, false);
    }

#pragma unroll
    for (int v = 0; v < 8; ++v) {
        const int M = v + 8 * half;
        float x = acc[v];
        if (act_out_silu) x = siluf(x);
        if (residual) x += residual[(r0 + M) * HID + n0 + n];
        out[(r0 + M) * HID + n0 + n] = x;
    }
}

// ---------------- GCN aggregation ----------------
__global__ void k_gcn_selfloop(const float* __restrict__ hw, const float* __restrict__ dinv,
                               float* __restrict__ out) {
    int i = blockIdx.x * blockDim.x + threadIdx.x;
    if (i < N_NODES * HID) {
        int node = i >> 6;
        float d = dinv[node];
        out[i] = hw[i] * d * d;
    }
}

__global__ void k_gcn_scatter(const float* __restrict__ hw, const int* __restrict__ ei,
                              const float* __restrict__ dinv, float* __restrict__ out) {
    int t = blockIdx.x * blockDim.x + threadIdx.x;
    if (t >= N_EDGES * 8) return;
    int e = t >> 3, j = t & 7;
    int s = ei[e], d = ei[N_EDGES + e];
    float w = dinv[s] * dinv[d];
    const float* src = hw + s * HID + j * 8;
    float* dst = out + d * HID + j * 8;
#pragma unroll
    for (int k = 0; k < 8; ++k) atomicAdd(&dst[k], src[k] * w);
}

// ---------------- fused EGCL edge pipeline: 16 edges / block, 4 waves ----------------
__global__ void __launch_bounds__(128) k_egcl_edges(
    const float* __restrict__ h, const float* __restrict__ coord,
    const int* __restrict__ ei, const float* __restrict__ eattr,
    const unsigned* __restrict__ eW1p, const float* __restrict__ eb1,
    const unsigned* __restrict__ eW2p, const float* __restrict__ eb2,
    const unsigned* __restrict__ cW1p, const float* __restrict__ cb1,
    const float* __restrict__ cW2,
    float* __restrict__ agg, float* __restrict__ coordAgg) {
    __shared__ __bf16 sIn[16][162];   // [h_i | h_j | r | ea0 ea1 | 0-pad to 160]
    __shared__ float  sDiff[16][32];
    __shared__ __bf16 sHid[16][66];
    __shared__ float  sM[16][64];
    __shared__ __bf16 sMb[16][66];
    __shared__ __bf16 sT[16][66];
    __shared__ float  sPhi[16];
    __shared__ int    sRow[16], sCol[16];

    const int t  = threadIdx.x;
    const int e0 = blockIdx.x * 16;
    if (t < 16) { sRow[t] = ei[e0 + t]; sCol[t] = ei[N_EDGES + e0 + t]; }
    __syncthreads();

    const int e = t >> 3, j = t & 7;  // 8 threads per edge
    const int row = sRow[e], col = sCol[e];
    {
        const float* hr = h + row * HID;
        const float* hc = h + col * HID;
        unsigned* rp = (unsigned*)&sIn[e][0];
#pragma unroll
        for (int kk = 0; kk < 4; ++kk) {
            int k = j * 2 + kk * 16;
            rp[k >> 1]        = pack2bf(hr[k], hr[k + 1]);
            rp[(64 + k) >> 1] = pack2bf(hc[k], hc[k + 1]);
        }
    }
    float r = 0.0f;
    {
        const float* cr = coord + row * NFEAT;
        const float* cc = coord + col * NFEAT;
#pragma unroll
        for (int dd = 0; dd < 4; ++dd) {
            int d = j + dd * 8;
            float cdv = cr[d] - cc[d];
            sDiff[e][d] = cdv;
            r += cdv * cdv;
        }
    }
    r += __shfl_xor(r, 1);
    r += __shfl_xor(r, 2);
    r += __shfl_xor(r, 4);
    if (j == 0) {
        sIn[e][128] = (__bf16)r;
        sIn[e][129] = (__bf16)eattr[(e0 + e) * 2 + 0];
        sIn[e][130] = (__bf16)eattr[(e0 + e) * 2 + 1];
    }
    for (int k = 131 + j; k < 162; k += 8) sIn[e][k] = (__bf16)0.0f;
    __syncthreads();

    const int lane = t & 31;
    const int wv   = t >> 5;
    const int n0   = wv * 16;
    const int half = lane >> 4, n = lane & 15;

    // GEMM1: e_in[16x131] @ eW1 + b1 -> silu -> sHid
    {
        v8f acc; const float b = eb1[n0 + n];
#pragma unroll
        for (int v = 0; v < 8; ++v) acc[v] = b;
#pragma unroll
        for (int fc = 0; fc < 5; ++fc) {
            v16bf a  = load_a_lds(&sIn[0][0], 162, fc * 32, lane);
            v16bf bm = load_b_packed(eW1p, fc, wv, lane);
            acc = __builtin_amdgcn_wmma_f32_16x16x32_bf16(false, a, false, bm, (short)0, acc, false, false);
        }
#pragma unroll
        for (int v = 0; v < 8; ++v) sHid[v + 8 * half][n0 + n] = (__bf16)siluf(acc[v]);
    }
    __syncthreads();

    // GEMM2: sHid @ eW2 + b2 -> silu -> m (f32 + bf16 copies)
    {
        v8f acc; const float b = eb2[n0 + n];
#pragma unroll
        for (int v = 0; v < 8; ++v) acc[v] = b;
#pragma unroll
        for (int fc = 0; fc < 2; ++fc) {
            v16bf a  = load_a_lds(&sHid[0][0], 66, fc * 32, lane);
            v16bf bm = load_b_packed(eW2p, fc, wv, lane);
            acc = __builtin_amdgcn_wmma_f32_16x16x32_bf16(false, a, false, bm, (short)0, acc, false, false);
        }
#pragma unroll
        for (int v = 0; v < 8; ++v) {
            float mv = siluf(acc[v]);
            sM[v + 8 * half][n0 + n]  = mv;
            sMb[v + 8 * half][n0 + n] = (__bf16)mv;
        }
    }
    __syncthreads();

    // GEMM3: m @ cW1 + cb1 -> silu -> sT
    {
        v8f acc; const float b = cb1[n0 + n];
#pragma unroll
        for (int v = 0; v < 8; ++v) acc[v] = b;
#pragma unroll
        for (int fc = 0; fc < 2; ++fc) {
            v16bf a  = load_a_lds(&sMb[0][0], 66, fc * 32, lane);
            v16bf bm = load_b_packed(cW1p, fc, wv, lane);
            acc = __builtin_amdgcn_wmma_f32_16x16x32_bf16(false, a, false, bm, (short)0, acc, false, false);
        }
#pragma unroll
        for (int v = 0; v < 8; ++v) sT[v + 8 * half][n0 + n] = (__bf16)siluf(acc[v]);
    }
    __syncthreads();

    // phi_x = sT @ cW2 (64 -> 1, no bias)
    float ps = 0.0f;
#pragma unroll
    for (int k = 0; k < 8; ++k) ps += (float)sT[e][j * 8 + k] * cW2[j * 8 + k];
    ps += __shfl_xor(ps, 1);
    ps += __shfl_xor(ps, 2);
    ps += __shfl_xor(ps, 4);
    if (j == 0) sPhi[e] = ps;
    __syncthreads();

    // scatter: coordAgg[row] += diff * phi ; agg[row] += m
    const float phi = sPhi[e];
#pragma unroll
    for (int dd = 0; dd < 4; ++dd) {
        int d = j + dd * 8;
        atomicAdd(&coordAgg[row * NFEAT + d], sDiff[e][d] * phi);
    }
#pragma unroll
    for (int k = 0; k < 8; ++k) atomicAdd(&agg[row * HID + j * 8 + k], sM[e][j * 8 + k]);
}

__global__ void k_coord_update(float* __restrict__ coord, const float* __restrict__ coordAgg,
                               const float* __restrict__ cnt) {
    int i = blockIdx.x * blockDim.x + threadIdx.x;
    if (i < N_NODES * NFEAT) {
        int node = i >> 5;
        coord[i] += coordAgg[i] / cnt[node];
    }
}

extern "C" void kernel_launch(void* const* d_in, const int* in_sizes, int n_in,
                              void* d_out, int out_size, void* d_ws, size_t ws_size,
                              hipStream_t stream) {
    (void)in_sizes; (void)n_in; (void)out_size; (void)ws_size;
    const float* x     = (const float*)d_in[0];
    const int*   ei    = (const int*)d_in[1];
    const float* eattr = (const float*)d_in[2];
    const float* gcnW[3] = {(const float*)d_in[3], (const float*)d_in[5], (const float*)d_in[7]};
    const float* gcnB[3] = {(const float*)d_in[4], (const float*)d_in[6], (const float*)d_in[8]};

    float* ws    = (float*)d_ws;
    float* dinv  = ws; ws += N_NODES;          // holds deg, then dinv
    float* cnt   = ws; ws += N_NODES;
    float* hA    = ws; ws += N_NODES * HID;
    float* hB    = ws; ws += N_NODES * HID;
    float* agg   = ws; ws += N_NODES * HID;
    float* coord = ws; ws += N_NODES * NFEAT;
    float* cAgg  = ws; ws += N_NODES * NFEAT;

    // packed-weight region (bf16 B-fragments), 1024 dwords per K-chunk
    unsigned* pk = (unsigned*)ws;
    unsigned* gcnWp[3];
    unsigned *eW1p[3], *eW2p[3], *cW1p[3], *nW1p[3], *nW2p[3];
    {
        int gcnKC[3] = {1, 2, 2};
        for (int l = 0; l < 3; ++l) { gcnWp[l] = pk; pk += gcnKC[l] * 1024; }
        for (int l = 0; l < 3; ++l) {
            eW1p[l] = pk; pk += 5 * 1024;
            eW2p[l] = pk; pk += 2 * 1024;
            cW1p[l] = pk; pk += 2 * 1024;
            nW1p[l] = pk; pk += 4 * 1024;
            nW2p[l] = pk; pk += 2 * 1024;
        }
    }

    const int TPB = 256;
    // ---- pack all weight matrices into WMMA B-fragment layout ----
    {
        int gcnK[3] = {32, 64, 64}, gcnKC[3] = {1, 2, 2};
        for (int l = 0; l < 3; ++l)
            k_pack_weights<<<(gcnKC[l] * 1024 + TPB - 1) / TPB, TPB, 0, stream>>>(gcnW[l], gcnK[l], gcnKC[l], gcnWp[l]);
        for (int l = 0; l < 3; ++l) {
            const int b = 9 + 11 * l;
            k_pack_weights<<<(5 * 1024 + TPB - 1) / TPB, TPB, 0, stream>>>((const float*)d_in[b + 0], 131, 5, eW1p[l]);
            k_pack_weights<<<(2 * 1024 + TPB - 1) / TPB, TPB, 0, stream>>>((const float*)d_in[b + 2], 64, 2, eW2p[l]);
            k_pack_weights<<<(2 * 1024 + TPB - 1) / TPB, TPB, 0, stream>>>((const float*)d_in[b + 8], 64, 2, cW1p[l]);
            k_pack_weights<<<(4 * 1024 + TPB - 1) / TPB, TPB, 0, stream>>>((const float*)d_in[b + 4], 128, 4, nW1p[l]);
            k_pack_weights<<<(2 * 1024 + TPB - 1) / TPB, TPB, 0, stream>>>((const float*)d_in[b + 6], 64, 2, nW2p[l]);
        }
    }

    // ---- degree / count / coord init ----
    k_fill<<<(N_NODES + TPB - 1) / TPB, TPB, 0, stream>>>(dinv, 1.0f, N_NODES);  // self loop
    k_fill<<<(N_NODES + TPB - 1) / TPB, TPB, 0, stream>>>(cnt, 0.0f, N_NODES);
    k_count<<<(N_EDGES + TPB - 1) / TPB, TPB, 0, stream>>>(ei, dinv, cnt);
    k_finalize<<<(N_NODES + TPB - 1) / TPB, TPB, 0, stream>>>(dinv, cnt);
    k_copy<<<(N_NODES * NFEAT + TPB - 1) / TPB, TPB, 0, stream>>>(coord, x, N_NODES * NFEAT);

    const int NT = N_NODES / 16;  // 3125 row tiles
    const int ET = N_EDGES / 16;  // 25000 edge tiles

    // ---- GCN encoder ----
    for (int l = 0; l < 3; ++l) {
        if (l == 0)
            k_rowtile_gemm<1><<<NT, 128, 0, stream>>>(x, NFEAT, nullptr, 0, gcnWp[0], gcnB[0],
                                                      nullptr, hB, 0, 0);
        else
            k_rowtile_gemm<2><<<NT, 128, 0, stream>>>(hA, HID, nullptr, 0, gcnWp[l], gcnB[l],
                                                      nullptr, hB, 1, 0);
        k_gcn_selfloop<<<(N_NODES * HID + TPB - 1) / TPB, TPB, 0, stream>>>(hB, dinv, hA);
        k_gcn_scatter<<<(N_EDGES * 8 + TPB - 1) / TPB, TPB, 0, stream>>>(hB, ei, dinv, hA);
    }

    // ---- EGCL stack ----
    for (int l = 0; l < 3; ++l) {
        const int b = 9 + 11 * l;
        const float* eb1 = (const float*)d_in[b + 1];
        const float* eb2 = (const float*)d_in[b + 3];
        const float* nb1 = (const float*)d_in[b + 5];
        const float* nb2 = (const float*)d_in[b + 7];
        const float* cb1 = (const float*)d_in[b + 9];
        const float* cW2 = (const float*)d_in[b + 10];

        k_fill<<<(N_NODES * HID + TPB - 1) / TPB, TPB, 0, stream>>>(agg, 0.0f, N_NODES * HID);
        k_fill<<<(N_NODES * NFEAT + TPB - 1) / TPB, TPB, 0, stream>>>(cAgg, 0.0f, N_NODES * NFEAT);
        k_egcl_edges<<<ET, 128, 0, stream>>>(hA, coord, ei, eattr, eW1p[l], eb1, eW2p[l], eb2,
                                             cW1p[l], cb1, cW2, agg, cAgg);
        k_coord_update<<<(N_NODES * NFEAT + TPB - 1) / TPB, TPB, 0, stream>>>(coord, cAgg, cnt);
        // node MLP: hB = silu([h|agg] @ nW1 + nb1) ; h = h + hB @ nW2 + nb2
        k_rowtile_gemm<4><<<NT, 128, 0, stream>>>(hA, HID, agg, HID, nW1p[l], nb1, nullptr, hB, 0, 1);
        float* outp = (l == 2) ? (float*)d_out : hA;
        k_rowtile_gemm<2><<<NT, 128, 0, stream>>>(hB, HID, nullptr, 0, nW2p[l], nb2, hA, outp, 0, 0);
    }
}
